// LiquidCell_35021163331819
// MI455X (gfx1250) — compile-verified, matrix operand order checked by reading, not run on hardware
//
#include <hip/hip_runtime.h>
#include <hip/hip_bf16.h>

typedef __bf16 bf16_t;
typedef __attribute__((ext_vector_type(16))) __bf16 v16bf;
typedef __attribute__((ext_vector_type(8)))  float  v8f;
typedef __attribute__((ext_vector_type(4)))  unsigned int u32x4;
typedef __attribute__((ext_vector_type(8)))  int i32x8;
typedef __attribute__((ext_vector_type(4)))  int i32x4;

#define I_DIM   512
#define H_DIM   512
#define K_DIM   1024            // I_DIM + H_DIM (GEMM K)
#define N_DIM   1024            // 512 f cols + 512 tau cols (GEMM N)
#define BATCHN  65536
#define MTILE   64              // rows per workgroup
#define BLOCKG  512             // 16 waves per workgroup
#define FRAG_BYTES 1024         // one 16x32 / 32x16 bf16 fragment: 32 lanes * 32B
#define A_LDS_BYTES (4 * 32 * FRAG_BYTES)   // 4 row-groups x 32 k-steps = 128KB
#define B_PANEL_BYTES (64 * FRAG_BYTES)     // one k-step x 64 n-tiles   = 64KB

// K index inside a 16x32 (A) or 32x16 (B) bf16 fragment, per CDNA5 ISA layout:
// lanes 0-15 hold K = {0..7, 16..23}, lanes 16-31 hold K = {8..15, 24..31};
// dwords 0..3 -> low-K half, dwords 4..7 -> high-K half, 2 elems per dword.
__device__ __forceinline__ int frag_k_local(int half, int i) {
    int d = i >> 1, p = i & 1;
    int base = (d < 4) ? 0 : 16;
    int dd   = (d < 4) ? d : d - 4;
    return base + (half ? 8 : 0) + dd * 2 + p;
}

// ---------------------------------------------------------------------------
// TDM: 1-D contiguous copy global -> LDS, n8 x 8-byte elements.
// D# per CDNA5 ISA ch.8: group0 = {count|flags, lds_addr, global_addr_lo,
// global_addr_hi|type=2}; group1 carries data_size / dims / strides.
// Issued by one wave; tracked with TENSORcnt.
// ---------------------------------------------------------------------------
__device__ __forceinline__ void tdm_copy_1d(void* lds_dst, const void* gsrc,
                                            unsigned n8) {
    unsigned lds_off = (unsigned)(size_t)lds_dst;     // low 32 bits = LDS offset
    unsigned long long ga = (unsigned long long)(size_t)gsrc;
    u32x4 g0;
    g0.x = 1u;                                        // count=1, user descriptor
    g0.y = lds_off;                                   // lds_addr (bytes)
    g0.z = (unsigned)(ga & 0xffffffffu);              // global_addr[31:0]
    g0.w = (unsigned)((ga >> 32) & 0x01ffffffu) | (2u << 30);  // addr hi | type=2
    i32x8 g1;
    g1[0] = (int)(3u << 16);                          // wg_mask=0, data_size=3 (8B)
    g1[1] = (int)((n8 & 0xffffu) << 16);              // tensor_dim0[15:0]
    g1[2] = (int)(((n8 >> 16) & 0xffffu) | (1u << 16)); // dim0[31:16] | tensor_dim1=1
    g1[3] = (int)((n8 & 0xffffu) << 16);              // tile_dim0 = n8
    g1[4] = 0;                                        // tile_dim1/2 = 0 (unused)
    g1[5] = (int)n8;                                  // tensor_dim0_stride
    g1[6] = 0;
    g1[7] = 0;
    i32x4 gz = {0, 0, 0, 0};
#if __clang_major__ >= 23
    i32x8 gz8 = {0, 0, 0, 0, 0, 0, 0, 0};
    __builtin_amdgcn_tensor_load_to_lds(g0, g1, gz, gz, gz8, 0);
#else
    __builtin_amdgcn_tensor_load_to_lds(g0, g1, gz, gz, 0);
#endif
}

// ---------------------------------------------------------------------------
// Kernel 1: pack [W_in|W_rec ; W_tau] -> bf16 in WMMA B-fragment order.
// Global layout: fragment id = kt*64 + nt (one k-step panel is contiguous);
// element offset inside fragment = lane*16 + i.
// ---------------------------------------------------------------------------
__global__ __launch_bounds__(256)
void pack_weights(const float* __restrict__ W_in, const float* __restrict__ W_rec,
                  const float* __restrict__ W_tau, bf16_t* __restrict__ Bw) {
    int D = blockIdx.x * blockDim.x + threadIdx.x;     // dword (bf16-pair) index
    if (D >= (N_DIM * K_DIM) / 2) return;
    int E    = D << 1;
    int frag = E >> 9;          // / 512 elems per fragment
    int r    = E & 511;
    int lane = r >> 4;
    int i    = r & 15;
    int kt   = frag >> 6;       // 0..31 (K step of 32)
    int nt   = frag & 63;       // 0..63 (N tile of 16)
    int hl   = lane >> 4, nl = lane & 15;
    int n    = nt * 16 + nl;                           // 0..1023
    int k    = kt * 32 + frag_k_local(hl, i);          // even; pair is k,k+1

    float2 w;
    if (n < H_DIM) {            // rows 0..511: [W_in | W_rec]
        if (k < I_DIM) w = *(const float2*)(W_in  + (size_t)n * I_DIM + k);
        else           w = *(const float2*)(W_rec + (size_t)n * H_DIM + (k - I_DIM));
    } else {                    // rows 512..1023: W_tau (already 1024 wide)
        w = *(const float2*)(W_tau + (size_t)(n - H_DIM) * K_DIM + k);
    }
    union { bf16_t b[2]; unsigned u; } cv;
    cv.b[0] = (bf16_t)w.x; cv.b[1] = (bf16_t)w.y;
    ((unsigned*)Bw)[D] = cv.u;
}

// ---------------------------------------------------------------------------
// One GEMM k-step: 1 A fragment x 16 B fragments -> 16 WMMAs.
// ---------------------------------------------------------------------------
__device__ __forceinline__ void kstep(const bf16_t* smA, const bf16_t* cur,
                                      int mg, int nc, int kt, int lane,
                                      v8f accF[8], v8f accT[8]) {
    const v16bf a = *(const v16bf*)(smA + ((mg * 32 + kt) << 9) + lane * 16);
    v16bf bfr[8];
    #pragma unroll
    for (int j = 0; j < 8; ++j)          // 8 f-col fragments
        bfr[j] = *(const v16bf*)(cur + ((nc * 8 + j) << 9) + lane * 16);
    #pragma unroll
    for (int j = 0; j < 8; ++j)
        accF[j] = __builtin_amdgcn_wmma_f32_16x16x32_bf16(
            false, a, false, bfr[j], (short)0, accF[j], false, false);
    #pragma unroll
    for (int j = 0; j < 8; ++j)          // 8 tau-col fragments
        bfr[j] = *(const v16bf*)(cur + ((32 + nc * 8 + j) << 9) + lane * 16);
    #pragma unroll
    for (int j = 0; j < 8; ++j)
        accT[j] = __builtin_amdgcn_wmma_f32_16x16x32_bf16(
            false, a, false, bfr[j], (short)0, accT[j], false, false);
}

// ---------------------------------------------------------------------------
// Kernel 2: fused GEMM + elementwise epilogue.
// out[0 .. B*H)        <- y = h + DT*(f - h)/tau   (pre-layernorm)
// out[B*H .. 2*B*H)    <- tau
// ---------------------------------------------------------------------------
__global__ __launch_bounds__(BLOCKG)
void liquid_gemm(const float* __restrict__ x, const float* __restrict__ hh,
                 const float* __restrict__ b_in, const float* __restrict__ b_tau,
                 const bf16_t* __restrict__ Bw, float* __restrict__ out) {
    extern __shared__ char smem[];
    bf16_t* smA  = (bf16_t*)smem;                           // 128KB: A fragments
    bf16_t* smB0 = (bf16_t*)(smem + A_LDS_BYTES);           // 64KB panel buf 0
    bf16_t* smB1 = (bf16_t*)(smem + A_LDS_BYTES + B_PANEL_BYTES); // 64KB buf 1

    const int tid  = threadIdx.x;
    const int lane = tid & 31;
    const int w    = tid >> 5;        // wave 0..15
    const int mg   = w & 3;           // row group (16 rows each)
    const int nc   = w >> 2;          // col chunk (128 f-cols + 128 tau-cols)
    const int rowBase = blockIdx.x * MTILE;

    // ---- Prologue: start DMA of B panel 0 while we stage A ----------------
    if (tid < 32) tdm_copy_1d(smB0, Bw, B_PANEL_BYTES / 8);

    // ---- Stage A: x|h rows -> bf16 fragments in LDS -----------------------
    // LDS fragment id = mg*32 + kt; 65536 elems total, 128 per thread.
    #pragma unroll 4
    for (int it = 0; it < 64; ++it) {
        int E    = tid * 128 + it * 2;
        int frag = E >> 9;
        int r    = E & 511;
        int al   = r >> 4;
        int i    = r & 15;
        int amg  = frag >> 5;
        int akt  = frag & 31;
        int hl   = al >> 4, ml = al & 15;
        int m    = rowBase + amg * 16 + ml;
        int k    = akt * 32 + frag_k_local(hl, i);
        float2 v;
        if (k < I_DIM) v = *(const float2*)(x  + (size_t)m * I_DIM + k);
        else           v = *(const float2*)(hh + (size_t)m * H_DIM + (k - I_DIM));
        union { bf16_t b[2]; unsigned u; } cv;
        cv.b[0] = (bf16_t)v.x; cv.b[1] = (bf16_t)v.y;
        ((unsigned*)smA)[E >> 1] = cv.u;
    }

    v8f accF[8], accT[8];
    #pragma unroll
    for (int j = 0; j < 8; ++j) {
        accF[j] = (v8f){0.f,0.f,0.f,0.f,0.f,0.f,0.f,0.f};
        accT[j] = (v8f){0.f,0.f,0.f,0.f,0.f,0.f,0.f,0.f};
    }

    if (tid < 32) __builtin_amdgcn_s_wait_tensorcnt(0);
    __syncthreads();

    // ---- Main loop: 32 k-steps, unrolled x2 so the TDM double-buffer ------
    // pointers are compile-time constants (no per-iteration cselect chains).
    #pragma unroll 1
    for (int kp = 0; kp < 16; ++kp) {
        const int kt0 = kp * 2;
        // even step: compute from buf0, DMA next panel into buf1
        if (tid < 32)
            tdm_copy_1d(smB1, (const char*)Bw + (size_t)(kt0 + 1) * B_PANEL_BYTES,
                        B_PANEL_BYTES / 8);
        kstep(smA, smB0, mg, nc, kt0, lane, accF, accT);
        if (tid < 32) __builtin_amdgcn_s_wait_tensorcnt(0);
        __syncthreads();

        // odd step: compute from buf1, DMA next panel into buf0
        if (kp + 1 < 16 && tid < 32)
            tdm_copy_1d(smB0, (const char*)Bw + (size_t)(kt0 + 2) * B_PANEL_BYTES,
                        B_PANEL_BYTES / 8);
        kstep(smA, smB1, mg, nc, kt0 + 1, lane, accF, accT);
        if (tid < 32) __builtin_amdgcn_s_wait_tensorcnt(0);
        __syncthreads();
    }

    // ---- Epilogue: tanh / sigmoid / liquid update, write y and tau --------
    // C/D layout: VGPR e, lanes 0-15 -> (M=e, N=lane); lanes 16-31 -> (M=e+8).
    const int nl = lane & 15;
    const int hf = lane >> 4;
    #pragma unroll
    for (int j = 0; j < 8; ++j) {
        int n = nc * 128 + j * 16 + nl;                 // hidden index 0..511
        float bi = b_in[n];
        float bt = b_tau[n];
        #pragma unroll
        for (int e = 0; e < 8; ++e) {
            int m = rowBase + mg * 16 + hf * 8 + e;
            float pre = accF[j][e] + bi;
            float tl  = accT[j][e] + bt;
            float hv  = hh[(size_t)m * H_DIM + n];
            // tanh via exp, sign-safe
            float t  = __expf(-2.f * fabsf(pre));
            float f  = copysignf((1.f - t) / (1.f + t), pre);
            float sg = 1.f / (1.f + __expf(-tl));
            float tau = 0.5f + 4.5f * sg;
            float y  = hv + 0.1f * (f - hv) / tau;
            out[(size_t)m * H_DIM + n] = y;                                  // pre-LN new_h
            out[(size_t)BATCHN * H_DIM + (size_t)m * H_DIM + n] = tau;       // tau output
        }
    }
}

// ---------------------------------------------------------------------------
// Kernel 3: in-place row layernorm over new_h. One wave32 per row.
// ---------------------------------------------------------------------------
__global__ __launch_bounds__(256)
void layernorm_rows(float* __restrict__ out, const float* __restrict__ gamma,
                    const float* __restrict__ beta) {
    const int lane = threadIdx.x & 31;
    const int wv   = threadIdx.x >> 5;
    const int row  = blockIdx.x * 8 + wv;
    float* y = out + (size_t)row * H_DIM + lane * 16;

    float4 v[4];
    float s = 0.f, s2 = 0.f;
    #pragma unroll
    for (int q = 0; q < 4; ++q) {
        v[q] = ((const float4*)y)[q];
        s  += v[q].x + v[q].y + v[q].z + v[q].w;
        s2 += v[q].x * v[q].x + v[q].y * v[q].y + v[q].z * v[q].z + v[q].w * v[q].w;
    }
    #pragma unroll
    for (int off = 16; off >= 1; off >>= 1) {
        s  += __shfl_xor(s,  off, 32);
        s2 += __shfl_xor(s2, off, 32);
    }
    float mu  = s * (1.f / (float)H_DIM);
    float var = s2 * (1.f / (float)H_DIM) - mu * mu;
    float rs  = rsqrtf(var + 1e-5f);

    const float* gp = gamma + lane * 16;
    const float* bp = beta  + lane * 16;
    #pragma unroll
    for (int q = 0; q < 4; ++q) {
        float4 g = ((const float4*)gp)[q];
        float4 b = ((const float4*)bp)[q];
        float4 o;
        o.x = (v[q].x - mu) * rs * g.x + b.x;
        o.y = (v[q].y - mu) * rs * g.y + b.y;
        o.z = (v[q].z - mu) * rs * g.z + b.z;
        o.w = (v[q].w - mu) * rs * g.w + b.w;
        ((float4*)y)[q] = o;
    }
}

// ---------------------------------------------------------------------------
extern "C" void kernel_launch(void* const* d_in, const int* in_sizes, int n_in,
                              void* d_out, int out_size, void* d_ws, size_t ws_size,
                              hipStream_t stream) {
    const float* x     = (const float*)d_in[0];
    const float* h     = (const float*)d_in[1];
    const float* W_in  = (const float*)d_in[2];
    const float* b_in  = (const float*)d_in[3];
    const float* W_rec = (const float*)d_in[4];
    const float* W_tau = (const float*)d_in[5];
    const float* b_tau = (const float*)d_in[6];
    const float* gamma = (const float*)d_in[7];
    const float* beta  = (const float*)d_in[8];
    bf16_t* Bw  = (bf16_t*)d_ws;          // 1024*1024 bf16 = 2MB scratch
    float*  out = (float*)d_out;          // [new_h (B*H) | tau (B*H)]

    pack_weights<<<(N_DIM * K_DIM / 2 + 255) / 256, 256, 0, stream>>>(W_in, W_rec, W_tau, Bw);
    liquid_gemm<<<BATCHN / MTILE, BLOCKG,
                  A_LDS_BYTES + 2 * B_PANEL_BYTES, stream>>>(
        x, h, b_in, b_tau, Bw, out);
    layernorm_rows<<<BATCHN / 8, 256, 0, stream>>>(out, gamma, beta);
}
